// Attn_4303557231022
// MI455X (gfx1250) — compile-verified, matrix operand order checked by reading, not run on hardware
//
#include <hip/hip_runtime.h>

typedef __attribute__((ext_vector_type(16))) __bf16 v16bf;
typedef __attribute__((ext_vector_type(8)))  float  v8f;
typedef unsigned int u32;

#define H 1024
#define S 2048
#define B 16
#define WAVES 8   // waves per block, one s-strip per wave

// round-to-nearest-even f32 -> bf16, packed pair (pure integer ops, always compiles)
__device__ __forceinline__ u32 pk_bf16(float lo, float hi) {
  u32 ul, uh;
  __builtin_memcpy(&ul, &lo, 4);
  __builtin_memcpy(&uh, &hi, 4);
  ul = ul + 0x7FFFu + ((ul >> 16) & 1u);
  uh = uh + 0x7FFFu + ((uh >> 16) & 1u);
  return (uh & 0xFFFF0000u) | (ul >> 16);
}

// gfx1250 has a hardware v_tanh_f32 (TRANS32 pipe, co-executes with XDL WMMA).
__device__ __forceinline__ float fast_tanh(float x) {
#if __has_builtin(__builtin_amdgcn_tanhf)
  return __builtin_amdgcn_tanhf(x);
#else
  return tanhf(x);
#endif
}

// ---------------------------------------------------------------------------
// Kernel 1: h_proj[b,k] = hidden[b,:] . attn_w[k, 0:H] + attn_b[k]   (tiny)
// ---------------------------------------------------------------------------
__global__ __launch_bounds__(256) void hproj_kernel(
    const float* __restrict__ hidden, const float* __restrict__ attn_w,
    const float* __restrict__ attn_b, float* __restrict__ hproj) {
  const int idx = blockIdx.x * 256 + threadIdx.x;   // idx = b*H + k
  const int b = idx >> 10;
  const int k = idx & (H - 1);
  const float* hp = hidden + (size_t)b * H;
  const float* wp = attn_w + (size_t)k * (2 * H);   // Wh row k
  float acc = attn_b[k];
  #pragma unroll 8
  for (int h = 0; h < H; ++h) acc = fmaf(hp[h], wp[h], acc);
  hproj[idx] = acc;
}

// ---------------------------------------------------------------------------
// Kernel 2 (main): fused enc@We^T -> tanh -> dot(score_w) per (s,b) row.
// One wave = one s value (16 rows = 16 batch entries, M dim of WMMA).
// ---------------------------------------------------------------------------
__global__ __launch_bounds__(256) void attn_energy_kernel(
    const float* __restrict__ enc,     // (S,B,H) f32
    const float* __restrict__ attn_w,  // (H,2H) f32; We = cols [H,2H)
    const float* __restrict__ score_w, // (1,H)
    const float* __restrict__ hproj,   // (B,H)
    float* __restrict__ energies) {    // (B,S)
  extern __shared__ __align__(32) char smem[];      // WAVES*32KB A + 32KB B
  const int tid  = threadIdx.x;
  const int wave = tid >> 5;
  const int lane = tid & 31;
  const int s    = blockIdx.x * WAVES + wave;       // this wave's strip
  const int kb   = (lane >= 16) ? 8 : 0;            // K-half per ISA A/B layout

  // ---- Stage A panel: enc rows (s, b=0..15) as bf16 WMMA A-fragments -------
  // Lane l<16 holds row M=l, K in {0..7,16..23}; lane l+16 holds K in {8..15,24..31}.
  {
    const float* arow = enc + ((size_t)s * B + (lane & 15)) * H;
    char* adst = smem + wave * 32768 + lane * 32;
    for (int ks = 0; ks < 32; ++ks) {
      u32 w[8];
      #pragma unroll
      for (int j = 0; j < 8; ++j) {
        const int K = kb + ((j < 4) ? (2 * j) : (16 + 2 * (j - 4)));
        const float2 f = *(const float2*)(arow + ks * 32 + K);
        w[j] = pk_bf16(f.x, f.y);
      }
      uint4* d = (uint4*)(adst + ks * 1024);
      d[0] = make_uint4(w[0], w[1], w[2], w[3]);
      d[1] = make_uint4(w[4], w[5], w[6], w[7]);
    }
  }
  __syncthreads();

  float part[8];
  #pragma unroll
  for (int r = 0; r < 8; ++r) part[r] = 0.f;

  const char* aLds = smem + wave * 32768 + lane * 32;
  const char* bLds = smem + 262144 + lane * 32;
  const int n  = lane & 15;
  const int mb = (lane < 16) ? 0 : 8;

  for (int nt = 0; nt < 64; ++nt) {                 // N-tiles over k (1024/16)
    __syncthreads();
    // ---- Stage B tile (We columns nt*16..+15, all K) cooperatively --------
    // slot = (kstep, laneSlot); B[kk][n] = attn_w[col_k][H + kk], contiguous in kk.
    #pragma unroll
    for (int q = 0; q < 4; ++q) {
      const int slot = tid + q * 256;               // 0..1023
      const int ks   = slot >> 5;
      const int l    = slot & 31;
      const int nn   = l & 15;
      const int kbb  = (l >= 16) ? 8 : 0;
      const float* src = attn_w + (size_t)(nt * 16 + nn) * (2 * H) + H + ks * 32 + kbb;
      const float4 a0 = *(const float4*)(src);
      const float4 a1 = *(const float4*)(src + 4);
      const float4 b0 = *(const float4*)(src + 16);
      const float4 b1 = *(const float4*)(src + 20);
      uint4* d = (uint4*)(smem + 262144 + ks * 1024 + l * 32);
      d[0] = make_uint4(pk_bf16(a0.x, a0.y), pk_bf16(a0.z, a0.w),
                        pk_bf16(a1.x, a1.y), pk_bf16(a1.z, a1.w));
      d[1] = make_uint4(pk_bf16(b0.x, b0.y), pk_bf16(b0.z, b0.w),
                        pk_bf16(b1.x, b1.y), pk_bf16(b1.z, b1.w));
      if (nt + 1 < 64) __builtin_prefetch(src + 16 * (2 * H), 0, 1); // next tile
    }
    __syncthreads();

    // ---- Preload epilogue operands: latency hides under the 32 WMMAs ------
    const int kcol = nt * 16 + n;
    const float sw = score_w[kcol];
    float hp[8];
    #pragma unroll
    for (int r = 0; r < 8; ++r) hp[r] = hproj[(mb + r) * H + kcol];

    // ---- 16x16 tile: accumulate over K = 1024 via 32 bf16 WMMAs -----------
    v8f c = {};
    #pragma unroll 4
    for (int ks = 0; ks < 32; ++ks) {
      v16bf a = *(const v16bf*)(aLds + ks * 1024);
      v16bf b = *(const v16bf*)(bLds + ks * 1024);
      c = __builtin_amdgcn_wmma_f32_16x16x32_bf16(
          /*neg_a=*/false, a, /*neg_b=*/false, b,
          /*c_mod=*/(short)0, c, /*reuse_a=*/false, /*reuse_b=*/false);
    }

    // ---- Epilogue: energy += tanh(C + h_proj) * score_w -------------------
    // C layout: VGPR r, lanes 0-15 -> (M=r, N=lane); lanes 16-31 -> (M=r+8).
    #pragma unroll
    for (int r = 0; r < 8; ++r) {
      part[r] += fast_tanh(c[r] + hp[r]) * sw;
    }
  }

  // ---- Reduce the 16 N-lanes of each row-half (stays inside each half) ----
  #pragma unroll
  for (int m = 8; m >= 1; m >>= 1) {
    #pragma unroll
    for (int r = 0; r < 8; ++r) part[r] += __shfl_xor(part[r], m, 32);
  }
  if (lane == 0 || lane == 16) {
    #pragma unroll
    for (int r = 0; r < 8; ++r)
      energies[(size_t)(mb + r) * S + s] = part[r];   // (B,S) layout
  }
}

// ---------------------------------------------------------------------------
// Kernel 3: mask + softmax over S, one block per batch row.
// ---------------------------------------------------------------------------
__global__ __launch_bounds__(256) void softmax_kernel(
    const float* __restrict__ energies, const int* __restrict__ seq_mask,
    float* __restrict__ out) {
  const int b = blockIdx.x;
  const int t = threadIdx.x;
  __shared__ float red[256];

  float v[8];
  float mx = -3.402823466e+38f;
  #pragma unroll
  for (int i = 0; i < 8; ++i) {
    const int s = t + i * 256;
    float e = energies[(size_t)b * S + s];
    if (seq_mask[(size_t)b * S + s] != 0) e = -1.0e12f;
    v[i] = e;
    mx = fmaxf(mx, e);
  }
  red[t] = mx; __syncthreads();
  for (int off = 128; off > 0; off >>= 1) {
    if (t < off) red[t] = fmaxf(red[t], red[t + off]);
    __syncthreads();
  }
  mx = red[0]; __syncthreads();

  float sum = 0.f;
  #pragma unroll
  for (int i = 0; i < 8; ++i) { v[i] = __expf(v[i] - mx); sum += v[i]; }
  red[t] = sum; __syncthreads();
  for (int off = 128; off > 0; off >>= 1) {
    if (t < off) red[t] += red[t + off];
    __syncthreads();
  }
  const float inv = 1.0f / red[0];
  #pragma unroll
  for (int i = 0; i < 8; ++i) out[(size_t)b * S + t + i * 256] = v[i] * inv;
}

// ---------------------------------------------------------------------------
extern "C" void kernel_launch(void* const* d_in, const int* in_sizes, int n_in,
                              void* d_out, int out_size, void* d_ws, size_t ws_size,
                              hipStream_t stream) {
  const float* hidden   = (const float*)d_in[0];   // (1,B,H)
  const float* enc      = (const float*)d_in[1];   // (S,B,H)
  const int*   seq_mask = (const int*)  d_in[2];   // (B,S)
  const float* attn_w   = (const float*)d_in[3];   // (H,2H)
  const float* attn_b   = (const float*)d_in[4];   // (H,)
  const float* score_w  = (const float*)d_in[5];   // (1,H)
  float* out = (float*)d_out;                      // (B,1,S)

  float* hproj    = (float*)d_ws;                  // B*H    = 16384 f32
  float* energies = (float*)d_ws + B * H;          // B*S    = 32768 f32

  hproj_kernel<<<(B * H) / 256, 256, 0, stream>>>(hidden, attn_w, attn_b, hproj);

  const size_t lds_bytes = WAVES * 32768 + 32768;  // 294912 B (<= 320KB WGP LDS)
  attn_energy_kernel<<<S / WAVES, 256, lds_bytes, stream>>>(
      enc, attn_w, score_w, hproj, energies);

  softmax_kernel<<<B, 256, 0, stream>>>(energies, seq_mask, out);
}
